// EdgeUpdaterModule_7851200217223
// MI455X (gfx1250) — matrix operand order repaired and checked by reading, not compiled, not run on hardware
//
#include <hip/hip_runtime.h>
#include <hip/hip_bf16.h>

typedef __attribute__((ext_vector_type(16))) __bf16 v16bf;
typedef __attribute__((ext_vector_type(8)))  __bf16 v8bf;
typedef __attribute__((ext_vector_type(8)))  float  v8f;

#define NE      524288
#define NDIM    128
#define HDIM    128
#define IN_DIM  384   // 2*ND + ED
#define TILE_E  64    // edges per block
#define MT      4     // 16-row M tiles per wave (B-fragment register reuse x4)

// ---------------------------------------------------------------------------
// Prep: convert W1 (384x128), Wg (384x128), W2 (128x128) fp32 -> bf16 in d_ws,
// pre-swizzled to the WMMA B-fragment layout:
//   fragment (nt, kt) occupies 512 bf16; lane L holds 16 consecutive bf16 with
//   K = kt*32 + (c/8)*16 + (L/16)*8 + c%8  (c = 0..15, maps to VGPR c/2 half c%2),
//   N = nt*16 + (L%16).
// ws layout: [W1bf: 49152][Wgbf: 49152][W2bf: 16384] bf16 elements.
// ---------------------------------------------------------------------------
__global__ __launch_bounds__(256)
void pack_weights_kernel(const float* __restrict__ W1,
                         const float* __restrict__ Wg,
                         const float* __restrict__ W2,
                         __bf16* __restrict__ ws)
{
    int t = blockIdx.x * 256 + threadIdx.x;
    const float* W; __bf16* dst; int KT; int e;
    if (t < 49152)        { W = W1; dst = ws;          KT = 12; e = t; }
    else if (t < 98304)   { W = Wg; dst = ws + 49152;  KT = 12; e = t - 49152; }
    else if (t < 114688)  { W = W2; dst = ws + 98304;  KT = 4;  e = t - 98304; }
    else return;

    int tile   = e >> 9;          // /512
    int within = e & 511;
    int nt = tile / KT;
    int kt = tile % KT;
    int L  = within >> 4;
    int c  = within & 15;
    int k  = kt * 32 + (c >> 3) * 16 + (L >> 4) * 8 + (c & 7);
    int n  = nt * 16 + (L & 15);
    dst[e] = (__bf16)W[k * HDIM + n];
}

// ---------------------------------------------------------------------------
// Main fused edge-update kernel. 256 threads = 8 waves; 64 edges per block.
// Wave w owns output columns [16w, 16w+16) for all 4 M tiles; each B fragment
// is loaded once and reused across the 4 M tiles (4x less L2 weight traffic).
// ---------------------------------------------------------------------------
__global__ __launch_bounds__(256)
void edge_update_kernel(const float*     __restrict__ x,
                        const long long* __restrict__ eidx,
                        const float*     __restrict__ ea,
                        const float*     __restrict__ b1,
                        const float*     __restrict__ g1,
                        const float*     __restrict__ be1,
                        const float*     __restrict__ b2,
                        const float*     __restrict__ bg,
                        const float*     __restrict__ gn,
                        const float*     __restrict__ bn,
                        const __bf16*    __restrict__ W1bf,
                        const __bf16*    __restrict__ Wgbf,
                        const __bf16*    __restrict__ W2bf,
                        float*           __restrict__ out)
{
    __shared__ __align__(16) __bf16 ldsA [TILE_E * IN_DIM]; // cat features bf16 (48KB)
    __shared__ __align__(16) __bf16 ldsA2[TILE_E * HDIM];   // gelu(ln(h1)) bf16 (16KB)
    __shared__ __align__(16) float  lds_h [TILE_E * HDIM];  // h1 f32 / out_pre f32 (32KB)
    __shared__ __align__(16) float  lds_ea[TILE_E * HDIM];  // edge_attr f32 (32KB)

    const int t     = threadIdx.x;
    const int row16 = t >> 4;     // 0..15, actual rows row16 + 16*i
    const int sub   = t & 15;     // 16 threads per row, 8 floats each
    const int w     = t >> 5;     // wave id 0..7 -> N tile
    const int lane  = t & 31;
    const int kgrp  = lane >> 4;  // C/D: rows r+8*kgrp within a 16-row tile
    const int nl    = lane & 15;  // C/D: column within tile

    const long long ebase = (long long)blockIdx.x * TILE_E;

    // ---- gather + fp32->bf16 convert into LDS (4 row-groups) --------------
    #pragma unroll
    for (int i = 0; i < MT; ++i) {
        const int row = row16 + 16 * i;
        const long long e = ebase + row;
        long long s = eidx[e];
        long long d = eidx[(long long)NE + e];
        const float4* ps = (const float4*)(x  + s * NDIM + sub * 8);
        const float4* pd = (const float4*)(x  + d * NDIM + sub * 8);
        const float4* pe = (const float4*)(ea + e * NDIM + sub * 8);
        float4 s0 = ps[0], s1 = ps[1];
        float4 d0 = pd[0], d1 = pd[1];
        float4 e0 = pe[0], e1 = pe[1];

        v8bf vs, vd, ve;
        vs[0]=(__bf16)s0.x; vs[1]=(__bf16)s0.y; vs[2]=(__bf16)s0.z; vs[3]=(__bf16)s0.w;
        vs[4]=(__bf16)s1.x; vs[5]=(__bf16)s1.y; vs[6]=(__bf16)s1.z; vs[7]=(__bf16)s1.w;
        vd[0]=(__bf16)d0.x; vd[1]=(__bf16)d0.y; vd[2]=(__bf16)d0.z; vd[3]=(__bf16)d0.w;
        vd[4]=(__bf16)d1.x; vd[5]=(__bf16)d1.y; vd[6]=(__bf16)d1.z; vd[7]=(__bf16)d1.w;
        ve[0]=(__bf16)e0.x; ve[1]=(__bf16)e0.y; ve[2]=(__bf16)e0.z; ve[3]=(__bf16)e0.w;
        ve[4]=(__bf16)e1.x; ve[5]=(__bf16)e1.y; ve[6]=(__bf16)e1.z; ve[7]=(__bf16)e1.w;

        *(v8bf*)(&ldsA[row * IN_DIM +       sub * 8]) = vs;
        *(v8bf*)(&ldsA[row * IN_DIM + 128 + sub * 8]) = vd;
        *(v8bf*)(&ldsA[row * IN_DIM + 256 + sub * 8]) = ve;
        ((float4*)(&lds_ea[row * HDIM + sub * 8]))[0] = e0;
        ((float4*)(&lds_ea[row * HDIM + sub * 8]))[1] = e1;
    }
    __syncthreads();

    // A-fragment loader: ISA 16-bit A 16x32 layout. lane m = lane%16,
    // elements 0..7  -> K = kt*32 + (lane/16)*8 + 0..7
    // elements 8..15 -> K = kt*32 + 16 + (lane/16)*8 + 0..7
    auto loadA = [&](const __bf16* base, int stride, int kt) -> v16bf {
        const __bf16* p = base + (lane & 15) * stride + kt * 32 + (lane >> 4) * 8;
        v8bf lo = *(const v8bf*)(p);
        v8bf hi = *(const v8bf*)(p + 16);
        return __builtin_shufflevector(lo, hi, 0,1,2,3,4,5,6,7,8,9,10,11,12,13,14,15);
    };
    // B-fragment loader: pre-swizzled, one 32B contiguous chunk per lane.
    auto loadB = [&](const __bf16* Wb, int KT, int kt) -> v16bf {
        return *(const v16bf*)(Wb + (((size_t)(w * KT + kt)) << 9) + (lane << 4));
    };

    // ---- GEMM1: h1 = cat @ W1 + b1 (B reused across 4 M tiles) ------------
    {
        v8f acc[MT];
        #pragma unroll
        for (int mt = 0; mt < MT; ++mt) acc[mt] = (v8f){0.f,0.f,0.f,0.f,0.f,0.f,0.f,0.f};
        #pragma unroll
        for (int kt = 0; kt < 12; ++kt) {
            v16bf b = loadB(W1bf, 12, kt);
            #pragma unroll
            for (int mt = 0; mt < MT; ++mt) {
                v16bf a = loadA(ldsA + mt * 16 * IN_DIM, IN_DIM, kt);
                acc[mt] = __builtin_amdgcn_wmma_f32_16x16x32_bf16(
                              false, a, false, b, (short)0, acc[mt], false, false);
            }
        }
        float bias = b1[w * 16 + nl];
        #pragma unroll
        for (int mt = 0; mt < MT; ++mt)
            #pragma unroll
            for (int r = 0; r < 8; ++r) {
                int m = mt * 16 + r + (kgrp << 3);
                lds_h[m * HDIM + w * 16 + nl] = acc[mt][r] + bias;
            }
    }
    __syncthreads();

    // ---- LayerNorm(g1,be1) + exact GELU -> bf16 A2 ------------------------
    #pragma unroll
    for (int i = 0; i < MT; ++i) {
        const int row = row16 + 16 * i;
        const float* hr = lds_h + row * HDIM + sub * 8;
        float v[8], s0 = 0.f, s1 = 0.f;
        #pragma unroll
        for (int j = 0; j < 8; ++j) { v[j] = hr[j]; s0 += v[j]; s1 += v[j] * v[j]; }
        #pragma unroll
        for (int off = 1; off < 16; off <<= 1) {
            s0 += __shfl_xor(s0, off, 16);
            s1 += __shfl_xor(s1, off, 16);
        }
        float mu   = s0 * (1.f / HDIM);
        float var  = s1 * (1.f / HDIM) - mu * mu;
        float rstd = rsqrtf(var + 1e-5f);
        v8bf o;
        #pragma unroll
        for (int j = 0; j < 8; ++j) {
            int c = sub * 8 + j;
            float xh = (v[j] - mu) * rstd * g1[c] + be1[c];
            float ge = 0.5f * xh * (1.f + erff(xh * 0.70710678118654752f));
            o[j] = (__bf16)ge;
        }
        *(v8bf*)(&ldsA2[row * HDIM + sub * 8]) = o;
    }
    __syncthreads();

    // ---- GEMM2: h2 = gelu @ W2 + b2 ---------------------------------------
    v8f acc2[MT];
    #pragma unroll
    for (int mt = 0; mt < MT; ++mt) acc2[mt] = (v8f){0.f,0.f,0.f,0.f,0.f,0.f,0.f,0.f};
    #pragma unroll
    for (int kt = 0; kt < 4; ++kt) {
        v16bf b = loadB(W2bf, 4, kt);
        #pragma unroll
        for (int mt = 0; mt < MT; ++mt) {
            v16bf a = loadA(ldsA2 + mt * 16 * HDIM, HDIM, kt);
            acc2[mt] = __builtin_amdgcn_wmma_f32_16x16x32_bf16(
                           false, a, false, b, (short)0, acc2[mt], false, false);
        }
    }
    // ---- GEMMg: gate pre-activation = cat @ Wg + bg -----------------------
    v8f accg[MT];
    #pragma unroll
    for (int mt = 0; mt < MT; ++mt) accg[mt] = (v8f){0.f,0.f,0.f,0.f,0.f,0.f,0.f,0.f};
    #pragma unroll
    for (int kt = 0; kt < 12; ++kt) {
        v16bf b = loadB(Wgbf, 12, kt);
        #pragma unroll
        for (int mt = 0; mt < MT; ++mt) {
            v16bf a = loadA(ldsA + mt * 16 * IN_DIM, IN_DIM, kt);
            accg[mt] = __builtin_amdgcn_wmma_f32_16x16x32_bf16(
                           false, a, false, b, (short)0, accg[mt], false, false);
        }
    }

    // ---- gate, residual: out_pre = h2 * sigmoid(g) + edge_attr ------------
    {
        float bias2 = b2[w * 16 + nl];
        float biasg = bg[w * 16 + nl];
        #pragma unroll
        for (int mt = 0; mt < MT; ++mt)
            #pragma unroll
            for (int r = 0; r < 8; ++r) {
                int m = mt * 16 + r + (kgrp << 3);
                float gate = 1.f / (1.f + expf(-(accg[mt][r] + biasg)));
                lds_h[m * HDIM + w * 16 + nl] =
                    (acc2[mt][r] + bias2) * gate + lds_ea[m * HDIM + w * 16 + nl];
            }
    }
    __syncthreads();

    // ---- final LayerNorm(gn,bn) -> global ---------------------------------
    #pragma unroll
    for (int i = 0; i < MT; ++i) {
        const int row = row16 + 16 * i;
        const long long e = ebase + row;
        const float* hr = lds_h + row * HDIM + sub * 8;
        float v[8], s0 = 0.f, s1 = 0.f;
        #pragma unroll
        for (int j = 0; j < 8; ++j) { v[j] = hr[j]; s0 += v[j]; s1 += v[j] * v[j]; }
        #pragma unroll
        for (int off = 1; off < 16; off <<= 1) {
            s0 += __shfl_xor(s0, off, 16);
            s1 += __shfl_xor(s1, off, 16);
        }
        float mu   = s0 * (1.f / HDIM);
        float var  = s1 * (1.f / HDIM) - mu * mu;
        float rstd = rsqrtf(var + 1e-5f);
        float4 o0, o1;
        int c = sub * 8;
        o0.x = (v[0]-mu)*rstd*gn[c+0] + bn[c+0];
        o0.y = (v[1]-mu)*rstd*gn[c+1] + bn[c+1];
        o0.z = (v[2]-mu)*rstd*gn[c+2] + bn[c+2];
        o0.w = (v[3]-mu)*rstd*gn[c+3] + bn[c+3];
        o1.x = (v[4]-mu)*rstd*gn[c+4] + bn[c+4];
        o1.y = (v[5]-mu)*rstd*gn[c+5] + bn[c+5];
        o1.z = (v[6]-mu)*rstd*gn[c+6] + bn[c+6];
        o1.w = (v[7]-mu)*rstd*gn[c+7] + bn[c+7];
        float4* po = (float4*)(out + e * HDIM + sub * 8);
        po[0] = o0;
        po[1] = o1;
    }
}

// ---------------------------------------------------------------------------
extern "C" void kernel_launch(void* const* d_in, const int* in_sizes, int n_in,
                              void* d_out, int out_size, void* d_ws, size_t ws_size,
                              hipStream_t stream) {
    const float*     x   = (const float*)d_in[0];
    const long long* ei  = (const long long*)d_in[1];
    const float*     ea  = (const float*)d_in[2];
    const float*     W1  = (const float*)d_in[3];
    const float*     b1  = (const float*)d_in[4];
    const float*     g1  = (const float*)d_in[5];
    const float*     be1 = (const float*)d_in[6];
    const float*     W2  = (const float*)d_in[7];
    const float*     b2  = (const float*)d_in[8];
    const float*     Wg  = (const float*)d_in[9];
    const float*     bg  = (const float*)d_in[10];
    const float*     gn  = (const float*)d_in[11];
    const float*     bn  = (const float*)d_in[12];

    __bf16* ws = (__bf16*)d_ws;   // 114688 bf16 = 229376 bytes used

    pack_weights_kernel<<<dim3(448), dim3(256), 0, stream>>>(W1, Wg, W2, ws);

    edge_update_kernel<<<dim3(NE / TILE_E), dim3(256), 0, stream>>>(
        x, ei, ea, b1, g1, be1, b2, bg, gn, bn,
        ws, ws + 49152, ws + 98304, (float*)d_out);
}
